// MoEExperts_17660905521868
// MI455X (gfx1250) — matrix operand order repaired and checked by reading, not compiled, not run on hardware
//
#include <hip/hip_runtime.h>
#include <cstdint>

#define TOKENS 2048
#define HIDDEN 2048
#define INTER  5632
#define NEXPERT 8
#define TOPK 2
#define NPAIRS (TOKENS*TOPK)

#define BM 128
#define BN 128
#define BK 32
#define MAXROWS 5120      // 4096 + 8*127 rounded up to /128
#define MAXMT   (MAXROWS/BM)

// ---- ws int-layout ----
#define WS_CNT 0
#define WS_OFF 8
#define WS_TOT 17
#define WS_MTE 32
#define WS_MTR 72
#define WS_TOK 128
#define WS_WGT 5248
#define WS_HOFF 65536

typedef __attribute__((ext_vector_type(16))) __bf16 v16bf;
typedef __attribute__((ext_vector_type(8)))  float  v8f;

union Frag16 { uint4 q[2]; v16bf v; };

// hardware packed f32->bf16 (RNE) : one VALU op instead of ~8
__device__ __forceinline__ uint32_t bf16_pk(float lo, float hi) {
  uint32_t r;
  asm("v_cvt_pk_bf16_f32 %0, %1, %2" : "=v"(r) : "v"(lo), "v"(hi));
  return r;
}
__device__ __forceinline__ uint16_t bf16_1(float f) {
  return (uint16_t)(bf16_pk(f, 0.f) & 0xFFFFu);
}

// async global->LDS b128 copy (ASYNCcnt-tracked, no VGPR round trip)
__device__ __forceinline__ void async_ld_b128(uint32_t lds_off, const void* gaddr) {
  asm volatile("global_load_async_to_lds_b128 %0, %1, off"
               :: "v"(lds_off), "v"(gaddr) : "memory");
}
__device__ __forceinline__ void wait_async0() {
  asm volatile("s_wait_asynccnt 0" ::: "memory");
}

// ---------------- routing ----------------
__global__ void moe_route1(const int* __restrict__ idx, int* __restrict__ wsI) {
  int t = threadIdx.x;
  if (t < NEXPERT) {
    int c = 0;
    for (int p = 0; p < NPAIRS; ++p) c += (idx[p] == t);
    wsI[WS_CNT + t] = c;
  }
  __syncthreads();
  if (t == 0) {
    int off = 0, mt = 0;
    wsI[WS_OFF] = 0;
    for (int e = 0; e < NEXPERT; ++e) {
      int c  = wsI[WS_CNT + e];
      int nt = (c + BM - 1) / BM;
      for (int j = 0; j < nt; ++j) { wsI[WS_MTE + mt] = e; wsI[WS_MTR + mt] = off + j * BM; ++mt; }
      off += nt * BM;
      wsI[WS_OFF + e + 1] = off;
    }
    wsI[WS_TOT] = mt;
  }
  for (int r = t; r < MAXROWS; r += blockDim.x) {
    wsI[WS_TOK + r] = -1;
    ((float*)wsI)[WS_WGT + r] = 0.f;
  }
}

__global__ void moe_route2(const int* __restrict__ idx, const float* __restrict__ wts,
                           int* __restrict__ wsI) {
  int p = blockIdx.x * blockDim.x + threadIdx.x;   // 0..4095
  int e = idx[p];
  int rank = 0;
  for (int q = 0; q < p; ++q) rank += (idx[q] == e);   // stable, deterministic slot
  int row = wsI[WS_OFF + e] + rank;
  wsI[WS_TOK + row] = p >> 1;
  ((float*)wsI)[WS_WGT + row] = wts[p];
}

// ---------------- gate/up GEMM + SwiGLU ----------------
__global__ __launch_bounds__(256)
void moe_gateup(const float* __restrict__ x, const float* __restrict__ w1,
                const float* __restrict__ w3, const int* __restrict__ wsI,
                uint16_t* __restrict__ hbuf) {
  int mt = blockIdx.y;
  if (mt >= wsI[WS_TOT]) return;
  int e    = wsI[WS_MTE + mt];
  int row0 = wsI[WS_MTR + mt];
  int n0   = blockIdx.x * BN;

  __shared__ __align__(16) uint32_t Ash [BM][BK / 2];   // [m][k-pair] bf16x2
  __shared__ __align__(16) uint32_t B1sh[BN][BK / 2];   // [n][k-pair]
  __shared__ __align__(16) uint32_t B3sh[BN][BK / 2];

  int tid = threadIdx.x;
  const float* W1 = w1 + (size_t)e * HIDDEN * INTER;
  const float* W3 = w3 + (size_t)e * HIDDEN * INTER;

  // hoisted A-row gather state (invariant over k loop)
  int    ak4 = (tid & 7) * 4;
  int    am[4];
  size_t abase[4];
  bool   aact[4];
#pragma unroll
  for (int i = 0; i < 4; ++i) {
    am[i] = i * 32 + (tid >> 3);
    int tok = wsI[WS_TOK + row0 + am[i]];
    aact[i]  = (tok >= 0);
    abase[i] = (size_t)(tok >= 0 ? tok : 0) * HIDDEN + ak4;
  }
  int bn4 = (tid & 31) * 4;
  int bkp = tid >> 5;

  v8f accG[4][2], accU[4][2];
#pragma unroll
  for (int i = 0; i < 4; ++i)
#pragma unroll
    for (int j = 0; j < 2; ++j) { accG[i][j] = {}; accU[i][j] = {}; }

  int wave = tid >> 5, lane = tid & 31;
  int wm = (wave & 1) * 64;
  int wn = (wave >> 1) * 32;
  int half = lane >> 4, l15 = lane & 15;

  for (int kk = 0; kk < HIDDEN; kk += BK) {
    __syncthreads();
#pragma unroll
    for (int i = 0; i < 4; ++i) {          // A tile 128x32
      float4 v = make_float4(0.f, 0.f, 0.f, 0.f);
      if (aact[i]) v = *(const float4*)(x + abase[i] + kk);
      uint2 d; d.x = bf16_pk(v.x, v.y); d.y = bf16_pk(v.z, v.w);
      *(uint2*)&Ash[am[i]][ak4 >> 1] = d;
    }
#pragma unroll
    for (int i = 0; i < 2; ++i) {          // B tiles 32x128 -> [n][k-pair]
      int kp = i * 8 + bkp;
      size_t g = (size_t)(kk + 2 * kp) * INTER + n0 + bn4;
      float4 p0 = *(const float4*)(W1 + g);
      float4 p1 = *(const float4*)(W1 + g + INTER);
      B1sh[bn4 + 0][kp] = bf16_pk(p0.x, p1.x);
      B1sh[bn4 + 1][kp] = bf16_pk(p0.y, p1.y);
      B1sh[bn4 + 2][kp] = bf16_pk(p0.z, p1.z);
      B1sh[bn4 + 3][kp] = bf16_pk(p0.w, p1.w);
      float4 q0 = *(const float4*)(W3 + g);
      float4 q1 = *(const float4*)(W3 + g + INTER);
      B3sh[bn4 + 0][kp] = bf16_pk(q0.x, q1.x);
      B3sh[bn4 + 1][kp] = bf16_pk(q0.y, q1.y);
      B3sh[bn4 + 2][kp] = bf16_pk(q0.z, q1.z);
      B3sh[bn4 + 3][kp] = bf16_pk(q0.w, q1.w);
    }
    __syncthreads();

    Frag16 bg[2], bu[2];
#pragma unroll
    for (int j = 0; j < 2; ++j) {
      const uint32_t* r1 = &B1sh[wn + j * 16 + l15][0];
      bg[j].q[0] = *(const uint4*)(r1 + half * 4);
      bg[j].q[1] = *(const uint4*)(r1 + 8 + half * 4);
      const uint32_t* r3 = &B3sh[wn + j * 16 + l15][0];
      bu[j].q[0] = *(const uint4*)(r3 + half * 4);
      bu[j].q[1] = *(const uint4*)(r3 + 8 + half * 4);
    }
#pragma unroll
    for (int i = 0; i < 4; ++i) {
      Frag16 a;
      const uint32_t* ar = &Ash[wm + i * 16 + l15][0];
      a.q[0] = *(const uint4*)(ar + half * 4);
      a.q[1] = *(const uint4*)(ar + 8 + half * 4);
#pragma unroll
      for (int j = 0; j < 2; ++j) {
        accG[i][j] = __builtin_amdgcn_wmma_f32_16x16x32_bf16(
            false, a.v, false, bg[j].v, (short)0, accG[i][j], false, false);
        accU[i][j] = __builtin_amdgcn_wmma_f32_16x16x32_bf16(
            false, a.v, false, bu[j].v, (short)0, accU[i][j], false, false);
      }
    }
  }
  // h = silu(g) * u, bf16 store
#pragma unroll
  for (int i = 0; i < 4; ++i)
#pragma unroll
    for (int j = 0; j < 2; ++j)
#pragma unroll
      for (int v = 0; v < 8; ++v) {
        int r = row0 + wm + i * 16 + v + half * 8;
        int c = n0 + wn + j * 16 + l15;
        float g = accG[i][j][v], u = accU[i][j][v];
        float hv = (g / (1.f + __expf(-g))) * u;
        hbuf[(size_t)r * INTER + c] = bf16_1(hv);
      }
}

// ---------------- down GEMM + weighted scatter ----------------
__global__ __launch_bounds__(256)
void moe_down(const uint16_t* __restrict__ hbuf, const float* __restrict__ w2,
              const int* __restrict__ wsI, float* __restrict__ out) {
  int mt = blockIdx.y;
  if (mt >= wsI[WS_TOT]) return;
  int e    = wsI[WS_MTE + mt];
  int row0 = wsI[WS_MTR + mt];
  int n0   = blockIdx.x * BN;

  __shared__ __align__(16) uint32_t Ash[BM][BK / 2];
  __shared__ __align__(16) uint32_t Bsh[BN][BK / 2];
  __shared__ int   tokS[BM];
  __shared__ float wS[BM];

  int tid = threadIdx.x;
  for (int r = tid; r < BM; r += 256) {
    tokS[r] = wsI[WS_TOK + row0 + r];
    wS[r]   = ((const float*)wsI)[WS_WGT + row0 + r];
  }

  const float* W2 = w2 + (size_t)e * INTER * HIDDEN;
  const uint32_t* h32 = (const uint32_t*)hbuf;

  // hoisted A-row state (bf16 h rows, dword units); async copies go straight to LDS
  int aq4 = (tid & 3) * 4;
  const uint32_t* hrow[2];
  uint32_t aldsoff[2];
#pragma unroll
  for (int i = 0; i < 2; ++i) {
    int m    = i * 64 + (tid >> 2);
    hrow[i]  = h32 + (size_t)(row0 + m) * (INTER / 2) + aq4;
    aldsoff[i] = (uint32_t)(uintptr_t)&Ash[m][aq4];   // LDS byte offset (low 32 of flat-LDS addr)
  }
  int bn4 = (tid & 31) * 4;
  int bkp = tid >> 5;

  v8f acc[4][2];
#pragma unroll
  for (int i = 0; i < 4; ++i)
#pragma unroll
    for (int j = 0; j < 2; ++j) acc[i][j] = {};

  int wave = tid >> 5, lane = tid & 31;
  int wm = (wave & 1) * 64;
  int wn = (wave >> 1) * 32;
  int half = lane >> 4, l15 = lane & 15;

  for (int kk = 0; kk < INTER; kk += BK) {
    __syncthreads();
#pragma unroll
    for (int i = 0; i < 2; ++i)            // A: async global->LDS b128
      async_ld_b128(aldsoff[i], (const void*)(hrow[i] + (kk >> 1)));
#pragma unroll
    for (int i = 0; i < 2; ++i) {          // B: w2 tile transposed, f32->bf16
      int kp = i * 8 + bkp;
      size_t g = (size_t)(kk + 2 * kp) * HIDDEN + n0 + bn4;
      float4 p0 = *(const float4*)(W2 + g);
      float4 p1 = *(const float4*)(W2 + g + HIDDEN);
      Bsh[bn4 + 0][kp] = bf16_pk(p0.x, p1.x);
      Bsh[bn4 + 1][kp] = bf16_pk(p0.y, p1.y);
      Bsh[bn4 + 2][kp] = bf16_pk(p0.z, p1.z);
      Bsh[bn4 + 3][kp] = bf16_pk(p0.w, p1.w);
    }
    wait_async0();                         // A-tile async writes visible
    __syncthreads();

    Frag16 b[2];
#pragma unroll
    for (int j = 0; j < 2; ++j) {
      const uint32_t* br = &Bsh[wn + j * 16 + l15][0];
      b[j].q[0] = *(const uint4*)(br + half * 4);
      b[j].q[1] = *(const uint4*)(br + 8 + half * 4);
    }
#pragma unroll
    for (int i = 0; i < 4; ++i) {
      Frag16 a;
      const uint32_t* ar = &Ash[wm + i * 16 + l15][0];
      a.q[0] = *(const uint4*)(ar + half * 4);
      a.q[1] = *(const uint4*)(ar + 8 + half * 4);
#pragma unroll
      for (int j = 0; j < 2; ++j)
        acc[i][j] = __builtin_amdgcn_wmma_f32_16x16x32_bf16(
            false, a.v, false, b[j].v, (short)0, acc[i][j], false, false);
    }
  }
#pragma unroll
  for (int i = 0; i < 4; ++i)
#pragma unroll
    for (int j = 0; j < 2; ++j)
#pragma unroll
      for (int v = 0; v < 8; ++v) {
        int rl = wm + i * 16 + v + half * 8;
        int tok = tokS[rl];
        if (tok >= 0) {
          int c = n0 + wn + j * 16 + l15;
          atomicAdd(&out[(size_t)tok * HIDDEN + c], acc[i][j][v] * wS[rl]);
        }
      }
}

extern "C" void kernel_launch(void* const* d_in, const int* in_sizes, int n_in,
                              void* d_out, int out_size, void* d_ws, size_t ws_size,
                              hipStream_t stream) {
  const float* x   = (const float*)d_in[0];
  const float* ew  = (const float*)d_in[1];
  const float* w1  = (const float*)d_in[2];
  const float* w2  = (const float*)d_in[3];
  const float* w3  = (const float*)d_in[4];
  const int*   idx = (const int*)d_in[5];
  float* out = (float*)d_out;

  int* wsI = (int*)d_ws;
  uint16_t* hbuf = (uint16_t*)((char*)d_ws + WS_HOFF);

  moe_route1<<<1, 256, 0, stream>>>(idx, wsI);
  moe_route2<<<NPAIRS / 256, 256, 0, stream>>>(idx, ew, wsI);
  hipMemsetAsync(d_out, 0, (size_t)out_size * sizeof(float), stream);
  moe_gateup<<<dim3(INTER / BN, MAXMT), 256, 0, stream>>>(x, w1, w3, wsI, hbuf);
  moe_down  <<<dim3(HIDDEN / BN, MAXMT), 256, 0, stream>>>(hbuf, w2, wsI, out);
}
